// GNNPretrain_59021440582061
// MI455X (gfx1250) — matrix operand order repaired and checked by reading, not compiled, not run on hardware
//
#include <hip/hip_runtime.h>

#define NF    128
#define DIM   256
#define HIDN  512
#define NLAYER 5
#define BN_EPS 1e-5f

typedef __attribute__((ext_vector_type(16))) __bf16 v16bf;
typedef __attribute__((ext_vector_type(8)))  float  v8f;

enum { OUT_F32_RELU = 0, OUT_F32_BIAS = 1, OUT_FRAG_RELU = 2 };

__device__ __forceinline__ unsigned short f2bf(float f) {
  unsigned int u = __float_as_uint(f);
  u += 0x7FFFu + ((u >> 16) & 1u);   // round-to-nearest-even to bf16
  return (unsigned short)(u >> 16);
}

// ---------------------------------------------------------------------------
// Swizzle fp32 row-major A (M x K) into WMMA A-fragment layout (bf16).
// Fragment (mblk, kblk) = 512 bf16 at lane*16 + i:
//   row = mblk*16 + (lane & 15)
//   K   = kblk*32 + ((i & 7) + ((i >> 3) << 4) + (lane >> 4) * 8)
// ---------------------------------------------------------------------------
__global__ void k_swizzleA(const float* __restrict__ A, unsigned short* __restrict__ out,
                           int M, int K) {
  int gid = blockIdx.x * blockDim.x + threadIdx.x;
  if (gid >= M * K) return;
  int i    = gid & 15;
  int lane = (gid >> 4) & 31;
  int frag = gid >> 9;
  int KB   = K >> 5;
  int kblk = frag % KB;
  int mblk = frag / KB;
  int row  = mblk * 16 + (lane & 15);
  int ko   = (i & 7) + ((i >> 3) << 4) + (lane >> 4) * 8;
  int k    = (kblk << 5) + ko;
  out[gid] = f2bf(A[(size_t)row * K + k]);
}

// ---------------------------------------------------------------------------
// Swizzle fp32 row-major B (K x Ncols) into WMMA B-fragment layout (bf16).
// Fragments ordered [nblk][kblk]; per lane element i:
//   k = kblk*32 + (i&7) + ((i>>3)<<4) + (lane>>4)*8 ; col = nblk*16 + (lane & 15)
// ---------------------------------------------------------------------------
__global__ void k_swizzleB(const float* __restrict__ B, unsigned short* __restrict__ out,
                           int K, int Ncols) {
  int gid = blockIdx.x * blockDim.x + threadIdx.x;
  if (gid >= K * Ncols) return;
  int i    = gid & 15;
  int lane = (gid >> 4) & 31;
  int frag = gid >> 9;
  int KB   = K >> 5;
  int kblk = frag % KB;
  int nblk = frag / KB;
  int k    = (kblk << 5) + (i & 7) + ((i >> 3) << 4) + (lane >> 4) * 8;
  int col  = (nblk << 4) + (lane & 15);
  out[gid] = f2bf(B[(size_t)k * Ncols + col]);
}

// ---------------------------------------------------------------------------
// Fragment GEMM with N-tile register blocking: 8 waves/block, each wave owns a
// 16 x (16*NT) output strip; one A-fragment load feeds NT WMMAs.
// MODE: OUT_F32_RELU / OUT_F32_BIAS -> fp32 row-major (+bias, opt. relu)
//       OUT_FRAG_RELU -> relu(bias+acc) emitted as bf16 directly in the NEXT
//       GEMM's A-fragment layout via a wave-private 16x16 LDS transpose tile.
// ---------------------------------------------------------------------------
template<int K, int NOUT, int NT, int MODE>
__global__ void k_gemm(const unsigned short* __restrict__ Af,
                       const unsigned short* __restrict__ Bf,
                       const float* __restrict__ bias,
                       void* __restrict__ Cv, int M) {
  constexpr int KB = K / 32;
  __shared__ __align__(16) unsigned short tbuf[8 * 256];  // 512B per wave

  int lane = threadIdx.x & 31;
  int wave = threadIdx.x >> 5;
  int mblk = blockIdx.x * 8 + wave;
  if (mblk * 16 >= M) return;           // wave-uniform: EXEC all-1 for WMMA
  int nb0  = blockIdx.y * NT;

  const unsigned short* ap = Af + (size_t)mblk * KB * 512 + lane * 16;
  const unsigned short* bp = Bf + (size_t)nb0 * KB * 512 + lane * 16;

  v8f acc[NT];
#pragma unroll
  for (int t = 0; t < NT; ++t) acc[t] = (v8f){};

  for (int kb = 0; kb < KB; ++kb) {
    v16bf a = *(const v16bf*)(ap + (size_t)kb * 512);
#pragma unroll
    for (int t = 0; t < NT; ++t) {
      v16bf b = *(const v16bf*)(bp + ((size_t)t * KB + kb) * 512);
      acc[t] = __builtin_amdgcn_wmma_f32_16x16x32_bf16(false, a, false, b, (short)0,
                                                       acc[t], false, false);
    }
  }

  int c15 = lane & 15;
  int g   = lane >> 4;
#pragma unroll
  for (int t = 0; t < NT; ++t) {
    int   nt  = nb0 + t;
    float bv  = bias[nt * 16 + c15];
    if (MODE == OUT_FRAG_RELU) {
      // D layout: lane owns column c15, rows g*8 .. g*8+7. Transpose through
      // wave-private LDS so each lane can emit its A-fragment 16B chunk.
      unsigned short* tb = tbuf + wave * 256;
#pragma unroll
      for (int r = 0; r < 8; ++r) {
        float v = fmaxf(acc[t][r] + bv, 0.f);
        tb[(r + 8 * g) * 16 + c15] = f2bf(v);
      }
      // same-wave DS ops are in order: RAW through LDS is safe, no barrier
      uint4 frag = *(const uint4*)(tb + c15 * 16 + g * 8);
      unsigned short* C = (unsigned short*)Cv;   // bf16 A-frag buffer, K_next = NOUT
      *(uint4*)(C + ((size_t)mblk * (NOUT / 32) + (nt >> 1)) * 512 +
                lane * 16 + (nt & 1) * 8) = frag;
    } else {
      float* C    = (float*)Cv;
      int   col   = nt * 16 + c15;
      int   rbase = mblk * 16 + g * 8;
#pragma unroll
      for (int r = 0; r < 8; ++r) {
        float v = acc[t][r] + bv;
        if (MODE == OUT_F32_RELU) v = fmaxf(v, 0.f);
        C[(size_t)(rbase + r) * NOUT + col] = v;
      }
    }
  }
}

// agg = h + self_emb  (self_emb = E1[l][4] + E2[l][0]), float4-vectorized
__global__ void k_init_agg4(const float4* __restrict__ h, const float* __restrict__ E1l,
                            const float* __restrict__ E2l, float4* __restrict__ agg, int n4) {
  int gid = blockIdx.x * blockDim.x + threadIdx.x;
  if (gid >= n4) return;
  int c = (gid & (DIM / 4 - 1)) * 4;
  float4 v = h[gid];
  float4 r;
  r.x = v.x + E1l[4 * DIM + c + 0] + E2l[c + 0];
  r.y = v.y + E1l[4 * DIM + c + 1] + E2l[c + 1];
  r.z = v.z + E1l[4 * DIM + c + 2] + E2l[c + 2];
  r.w = v.w + E1l[4 * DIM + c + 3] + E2l[c + 3];
  agg[gid] = r;
}

// agg[dst] += h[src] + E1[bt] + E2[bd]   (one block per edge, lane = channel)
__global__ void k_edge_scatter(const float* __restrict__ h,
                               const int* __restrict__ ei,   // (2, E)
                               const int* __restrict__ ea,   // (E, 2)
                               const float* __restrict__ E1l,
                               const float* __restrict__ E2l,
                               float* __restrict__ agg, int E) {
  int e = blockIdx.x;
  int c = threadIdx.x;
  int s  = ei[e];
  int d  = ei[E + e];
  int bt = ea[2 * e];
  int bd = ea[2 * e + 1];
  float v = h[(size_t)s * DIM + c] + E1l[bt * DIM + c] + E2l[bd * DIM + c];
  unsafeAtomicAdd(&agg[(size_t)d * DIM + c], v);
}

__global__ void k_zero(float* p, int n) {
  int g = blockIdx.x * blockDim.x + threadIdx.x;
  if (g < n) p[g] = 0.f;
}

// per-column sum / sumsq partials over 512-row chunks (coalesced b32 rows)
__global__ void k_colstats(const float* __restrict__ out, float* __restrict__ sum,
                           float* __restrict__ sumsq, int N) {
  int c  = threadIdx.x;                 // 256 lanes = 256 columns
  int r0 = blockIdx.x * 512;
  int r1 = r0 + 512; if (r1 > N) r1 = N;
  float s = 0.f, s2 = 0.f;
  for (int r = r0; r < r1; ++r) {
    float v = out[(size_t)r * DIM + c];
    s += v; s2 += v * v;
  }
  unsafeAtomicAdd(&sum[c], s);
  unsafeAtomicAdd(&sumsq[c], s2);
}

__global__ void k_bn_final(const float* __restrict__ sum, const float* __restrict__ sumsq,
                           const float* __restrict__ gamma, const float* __restrict__ beta,
                           float* __restrict__ scale, float* __restrict__ shift, int N) {
  int c = threadIdx.x;
  float m   = sum[c] / (float)N;
  float var = sumsq[c] / (float)N - m * m;
  float is  = rsqrtf(var + BN_EPS);
  float sc  = gamma[c] * is;
  scale[c] = sc;
  shift[c] = beta[c] - m * sc;
}

__global__ void k_bn_apply4(const float4* __restrict__ out, const float* __restrict__ scale,
                            const float* __restrict__ shift, float4* __restrict__ h, int n4) {
  int gid = blockIdx.x * blockDim.x + threadIdx.x;
  if (gid >= n4) return;
  int c = (gid & (DIM / 4 - 1)) * 4;
  float4 v = out[gid];
  float4 r;
  r.x = v.x * scale[c + 0] + shift[c + 0];
  r.y = v.y * scale[c + 1] + shift[c + 1];
  r.z = v.z * scale[c + 2] + shift[c + 2];
  r.w = v.w * scale[c + 3] + shift[c + 3];
  h[gid] = r;
}

extern "C" void kernel_launch(void* const* d_in, const int* in_sizes, int n_in,
                              void* d_out, int out_size, void* d_ws, size_t ws_size,
                              hipStream_t stream) {
  const float* x     = (const float*)d_in[0];
  const int*   ei    = (const int*)d_in[1];
  const int*   ea    = (const int*)d_in[2];
  const float* Wl    = (const float*)d_in[3];
  const float* bl    = (const float*)d_in[4];
  const float* W1    = (const float*)d_in[5];
  const float* b1    = (const float*)d_in[6];
  const float* W2    = (const float*)d_in[7];
  const float* b2    = (const float*)d_in[8];
  const float* E1    = (const float*)d_in[9];
  const float* E2    = (const float*)d_in[10];
  const float* gamma = (const float*)d_in[11];
  const float* beta  = (const float*)d_in[12];

  const int N = in_sizes[0] / NF;   // 100000 (multiple of 16)
  const int E = in_sizes[1] / 2;    // 400000

  // ---- workspace carve-up ----
  char*  ws  = (char*)d_ws;
  size_t off = 0;
  float* h    = (float*)(ws + off); off += (size_t)N * DIM * sizeof(float);
  float* agg  = (float*)(ws + off); off += (size_t)N * DIM * sizeof(float);  // also 'out'
  unsigned short* afragA = (unsigned short*)(ws + off);
  off += (size_t)N * DIM * sizeof(unsigned short);   // x-frags (K=128) / agg-frags (K=256)
  unsigned short* hfrag = (unsigned short*)(ws + off);
  off += (size_t)N * HIDN * sizeof(unsigned short);  // hid frags (K=512), written by GEMM1
  unsigned short* wlf = (unsigned short*)(ws + off); off += (size_t)NF     * DIM  * 2;
  unsigned short* w1f = (unsigned short*)(ws + off); off += (size_t)NLAYER * DIM  * HIDN * 2;
  unsigned short* w2f = (unsigned short*)(ws + off); off += (size_t)NLAYER * HIDN * DIM  * 2;
  float* colsum = (float*)(ws + off); off += DIM * sizeof(float);
  float* colsq  = (float*)(ws + off); off += DIM * sizeof(float);
  float* scale  = (float*)(ws + off); off += DIM * sizeof(float);
  float* shift  = (float*)(ws + off); off += DIM * sizeof(float);

  const int T = 256;
  const unsigned mgrid = (unsigned)((N / 16 + 7) / 8);
  const unsigned n4    = (unsigned)((size_t)N * DIM / 4);

  // ---- weight fragment swizzles ----
  k_swizzleB<<<(NF * DIM + T - 1) / T, T, 0, stream>>>(Wl, wlf, NF, DIM);
  for (int l = 0; l < NLAYER; ++l) {
    k_swizzleB<<<(DIM * HIDN + T - 1) / T, T, 0, stream>>>(
        W1 + (size_t)l * DIM * HIDN, w1f + (size_t)l * DIM * HIDN, DIM, HIDN);
    k_swizzleB<<<(HIDN * DIM + T - 1) / T, T, 0, stream>>>(
        W2 + (size_t)l * HIDN * DIM, w2f + (size_t)l * HIDN * DIM, HIDN, DIM);
  }

  // ---- input projection: h = relu(x @ Wl + bl) ----
  k_swizzleA<<<(unsigned)(((size_t)N * NF + T - 1) / T), T, 0, stream>>>(x, afragA, N, NF);
  {
    dim3 g(mgrid, (DIM / 16) / 8);   // NT=8
    k_gemm<NF, DIM, 8, OUT_F32_RELU><<<g, T, 0, stream>>>(afragA, wlf, bl, h, N);
  }

  for (int l = 0; l < NLAYER; ++l) {
    const float* E1l = E1 + (size_t)l * 6 * DIM;
    const float* E2l = E2 + (size_t)l * 3 * DIM;
    float* out = agg;   // alias: agg is dead once its fragments are built

    k_init_agg4<<<(n4 + T - 1) / T, T, 0, stream>>>((const float4*)h, E1l, E2l,
                                                    (float4*)agg, n4);
    k_edge_scatter<<<(unsigned)E, DIM, 0, stream>>>(h, ei, ea, E1l, E2l, agg, E);

    k_swizzleA<<<(unsigned)(((size_t)N * DIM + T - 1) / T), T, 0, stream>>>(agg, afragA,
                                                                            N, DIM);
    {
      dim3 g(mgrid, (HIDN / 16) / 8);  // NT=8; writes bf16 frags (K_next=512) directly
      k_gemm<DIM, HIDN, 8, OUT_FRAG_RELU><<<g, T, 0, stream>>>(
          afragA, w1f + (size_t)l * DIM * HIDN, b1 + (size_t)l * HIDN, hfrag, N);
    }
    {
      dim3 g(mgrid, (DIM / 16) / 8);   // NT=8
      k_gemm<HIDN, DIM, 8, OUT_F32_BIAS><<<g, T, 0, stream>>>(
          hfrag, w2f + (size_t)l * HIDN * DIM, b2 + (size_t)l * DIM, out, N);
    }

    // batch norm
    k_zero<<<(2 * DIM + T - 1) / T, T, 0, stream>>>(colsum, 2 * DIM);
    k_colstats<<<(unsigned)((N + 511) / 512), DIM, 0, stream>>>(out, colsum, colsq, N);
    k_bn_final<<<1, DIM, 0, stream>>>(colsum, colsq, gamma + (size_t)l * DIM,
                                      beta + (size_t)l * DIM, scale, shift, N);
    float* hnext = (l == NLAYER - 1) ? (float*)d_out : h;
    k_bn_apply4<<<(n4 + T - 1) / T, T, 0, stream>>>((const float4*)out, scale, shift,
                                                    (float4*)hnext, n4);
  }
}
